// SFNN_8005819040026
// MI455X (gfx1250) — matrix operand order repaired and checked by reading, not compiled, not run on hardware
//
#include <hip/hip_runtime.h>

#define N_NODES 1024
#define DH 32
#define DE 16
#define T_TYPES 8
#define ADIM 10
#define NOBS 64

typedef __attribute__((ext_vector_type(16))) __bf16 v16bf;
typedef __attribute__((ext_vector_type(8)))  float  v8f;

// Fast activations: hardware rcp/exp (and tanh trans-op if this clang exposes it).
__device__ __forceinline__ float sigmoid_fast(float x) {
    return __builtin_amdgcn_rcpf(1.0f + __expf(-x));
}
__device__ __forceinline__ float tanh_fast(float x) {
#if __has_builtin(__builtin_amdgcn_tanhf)
    return __builtin_amdgcn_tanhf(x);
#elif __has_builtin(__builtin_amdgcn_tanh_f32)
    return __builtin_amdgcn_tanh_f32(x);
#else
    float t = __expf(2.0f * x);                      // e^{2x}
    return 1.0f - 2.0f * __builtin_amdgcn_rcpf(t + 1.0f);
#endif
}

// y[n][d] = relu(sum_k h[n][k] * W_out[d][k] + b_out[d])
__global__ void k_node_out(const float* __restrict__ h, const float* __restrict__ W_out,
                           const float* __restrict__ b_out, float* __restrict__ y) {
    int idx = blockIdx.x * blockDim.x + threadIdx.x;
    if (idx >= N_NODES * DE) return;
    int n = idx >> 4, d = idx & 15;
    float acc = b_out[d];
    const float* hr = h + n * DH;
    const float* wr = W_out + d * DH;
#pragma unroll
    for (int k = 0; k < DH; ++k) acc += hr[k] * wr[k];
    y[idx] = fmaxf(acc, 0.0f);
}

// x[i][d] = y[i][d] * sum_j e[i][j][d] * A[i][j];  x[:NOBS,0] = obs
__global__ void k_aggregate(const float* __restrict__ e, const int* __restrict__ A,
                            const float* __restrict__ y, const float* __restrict__ obs,
                            float* __restrict__ x) {
    int i = blockIdx.x;
    int tid = threadIdx.x;          // 256 threads
    int d = tid & 15;
    int jg = tid >> 4;              // 16 j-groups
    const float* er = e + (size_t)i * N_NODES * DE;
    const int*   Ar = A + (size_t)i * N_NODES;
    float acc = 0.0f;
    for (int j = jg; j < N_NODES; j += 16)
        acc += er[(size_t)j * DE + d] * (float)Ar[j];
    __shared__ float red[256];
    red[tid] = acc;
    __syncthreads();
    if (tid < DE) {
        float s = 0.0f;
#pragma unroll
        for (int k = 0; k < 16; ++k) s += red[tid + (k << 4)];
        float val = y[i * DE + tid] * s;
        if (i < NOBS && tid == 0) val = obs[i];
        x[i * DE + tid] = val;
    }
}

// Per-node GRU, type-gathered weights. One block per node.
__global__ void k_node_gru(const float* __restrict__ x, const float* __restrict__ h,
                           const int* __restrict__ types,
                           const float* __restrict__ w_ih_n, const float* __restrict__ w_hh_n,
                           const float* __restrict__ b_n, const float* __restrict__ bn_n,
                           float* __restrict__ h_out) {
    int n = blockIdx.x;
    int tid = threadIdx.x;          // 128 threads
    int t = types[n];
    __shared__ float xv[DE], hv[DH], igs[3 * DH], hgs[3 * DH];
    if (tid < DE) xv[tid] = x[n * DE + tid];
    if (tid < DH) hv[tid] = h[n * DH + tid];
    __syncthreads();
    if (tid < 3 * DH) {
        int g = tid;
        const float* wi = w_ih_n + ((size_t)t * 3 * DH + g) * DE;
        const float* wh = w_hh_n + ((size_t)t * 3 * DH + g) * DH;
        float ia = b_n[t * 3 * DH + g];
        float ha = 0.0f;
#pragma unroll
        for (int d = 0; d < DE; ++d) ia += wi[d] * xv[d];
#pragma unroll
        for (int d = 0; d < DH; ++d) ha += wh[d] * hv[d];
        igs[g] = ia;
        hgs[g] = ha;
    }
    __syncthreads();
    if (tid < DH) {
        int d = tid;
        float r  = sigmoid_fast(igs[d] + hgs[d]);
        float z  = sigmoid_fast(igs[DH + d] + hgs[DH + d]);
        float nn = tanh_fast(igs[2 * DH + d] + r * (hgs[2 * DH + d] + bn_n[t * DH + d]));
        h_out[n * DH + d] = nn + z * (hv[d] - nn);
    }
}

// Crow[i][g] = sum_d w_ih_e[t_i][g][d]*y[i][d] + w_ih_e[t_i][g][2*DE]*r + b_e[t_i][g]
__global__ void k_edge_rowpre(const float* __restrict__ w_ih_e, const float* __restrict__ y,
                              const float* __restrict__ b_e, const float* __restrict__ r,
                              const int* __restrict__ types, float* __restrict__ Crow) {
    int i = blockIdx.x;
    int g = threadIdx.x;
    if (g >= 3 * DE) return;
    int t = types[i];
    const float* w  = w_ih_e + ((size_t)t * 3 * DE + g) * (2 * DE + 1);
    const float* yi = y + i * DE;
    float acc = b_e[t * 3 * DE + g] + w[2 * DE] * r[0];
#pragma unroll
    for (int d = 0; d < DE; ++d) acc += w[d] * yi[d];
    Crow[i * (3 * DE) + g] = acc;
}

// Edge GRU. Per row i, per 16-wide j-tile:
//   pre_r/z = E_tile @ W_hh^T + Y_tile @ W_ih[:,16:32]^T + c_i   (chained WMMAs, c via C operand)
//   hn      = E_tile @ W_hh^T[n-block]
//   ign     = Y_tile @ W_ih^T[n-block] + c_n
// then branch-free GRU gates and coalesced store. 6 WMMAs / tile, no gather tables.
__global__ void __launch_bounds__(128)
k_edge_gru(const float* __restrict__ e, const int* __restrict__ types,
           const float* __restrict__ w_hh_e, const float* __restrict__ w_ih_e,
           const float* __restrict__ bn_e, const float* __restrict__ Crow,
           const float* __restrict__ y, float* __restrict__ e_out) {
    const int i    = blockIdx.x;
    const int wave = threadIdx.x >> 5;
    const int lane = threadIdx.x & 31;
    const int hi   = lane >> 4;     // half-wave
    const int m    = lane & 15;     // column index within tile
    const int t    = types[i];

    // B fragments: B[K=d][N=g], K=16 padded to 32 with zeros.
    // Lanes 0-15: K=0..7 in halves 0..7, lanes 16-31: K=8..15 (N = lane%16); halves 8..15 zero.
    v16bf Bhh[3], Bih[3];
#pragma unroll
    for (int gb = 0; gb < 3; ++gb) {
        v16bf bh, bi;
#pragma unroll
        for (int hh = 0; hh < 16; ++hh) { bh[hh] = (__bf16)0.0f; bi[hh] = (__bf16)0.0f; }
        {
            const int g = gb * 16 + m;
            const float4* wr = (const float4*)(w_hh_e + ((size_t)t * 3 * DE + g) * DE + hi * 8);
#pragma unroll
            for (int q = 0; q < 2; ++q) {
                float4 f = wr[q];
                bh[4 * q + 0] = (__bf16)f.x;
                bh[4 * q + 1] = (__bf16)f.y;
                bh[4 * q + 2] = (__bf16)f.z;
                bh[4 * q + 3] = (__bf16)f.w;
            }
            const float* wi = w_ih_e + ((size_t)t * 3 * DE + g) * (2 * DE + 1) + DE + hi * 8;
#pragma unroll
            for (int hh = 0; hh < 8; ++hh) bi[hh] = (__bf16)wi[hh];
        }
        Bhh[gb] = bh;
        Bih[gb] = bi;
    }

    const float c0  = Crow[i * 48 + m];
    const float c1  = Crow[i * 48 + 16 + m];
    const float c2  = Crow[i * 48 + 32 + m];
    const float bnv = bn_e[t * DE + m];
    v8f c0v, c1v, c2v, zero8 = {};
#pragma unroll
    for (int q = 0; q < 8; ++q) { c0v[q] = c0; c1v[q] = c1; c2v[q] = c2; }

    const float* erow = e + (size_t)i * N_NODES * DE;
    float*       orow = e_out + (size_t)i * N_NODES * DE;

    for (int jt = wave; jt < N_NODES / 16; jt += 4) {
        const int j0 = jt * 16;
        // A fragments (16x32 bf16, K 16->32 zero-pad): lane holds row M = j0+m;
        // lanes 0-15: K=0..7 in halves 0..7, lanes 16-31: K=8..15; halves 8..15 zero.
        v16bf ae, ay;
        {
            const float4* ep = (const float4*)(erow + ((size_t)(j0 + m)) * DE + hi * 8);
            const float4* yp = (const float4*)(y + ((size_t)(j0 + m)) * DE + hi * 8);
            float4 e0 = ep[0], e1 = ep[1];
            float4 y0 = yp[0], y1 = yp[1];
            ae[0] = (__bf16)e0.x; ae[1] = (__bf16)e0.y; ae[2] = (__bf16)e0.z; ae[3] = (__bf16)e0.w;
            ae[4] = (__bf16)e1.x; ae[5] = (__bf16)e1.y; ae[6] = (__bf16)e1.z; ae[7] = (__bf16)e1.w;
            ay[0] = (__bf16)y0.x; ay[1] = (__bf16)y0.y; ay[2] = (__bf16)y0.z; ay[3] = (__bf16)y0.w;
            ay[4] = (__bf16)y1.x; ay[5] = (__bf16)y1.y; ay[6] = (__bf16)y1.z; ay[7] = (__bf16)y1.w;
#pragma unroll
            for (int hh = 8; hh < 16; ++hh) { ae[hh] = (__bf16)0.0f; ay[hh] = (__bf16)0.0f; }
        }

        // r/z gates: chained accumulation (hg + ig + c) fully inside the matrix unit.
        v8f pre_r = __builtin_amdgcn_wmma_f32_16x16x32_bf16(false, ae, false, Bhh[0], (short)0, c0v,   false, false);
        pre_r     = __builtin_amdgcn_wmma_f32_16x16x32_bf16(false, ay, false, Bih[0], (short)0, pre_r, false, false);
        v8f pre_z = __builtin_amdgcn_wmma_f32_16x16x32_bf16(false, ae, false, Bhh[1], (short)0, c1v,   false, false);
        pre_z     = __builtin_amdgcn_wmma_f32_16x16x32_bf16(false, ay, false, Bih[1], (short)0, pre_z, false, false);
        // n gate halves stay separate (r multiplies (hn + bn)).
        v8f hn8   = __builtin_amdgcn_wmma_f32_16x16x32_bf16(false, ae, false, Bhh[2], (short)0, zero8, false, false);
        v8f ign8  = __builtin_amdgcn_wmma_f32_16x16x32_bf16(false, ay, false, Bih[2], (short)0, c2v,   false, false);

        // Accumulator layout: element v of lane (hi,m) = D[j_local = v + 8*hi][g = m]
#pragma unroll
        for (int v = 0; v < 8; ++v) {
            const int j = j0 + v + 8 * hi;
            float eo = erow[(size_t)j * DE + m];
            float rg = sigmoid_fast(pre_r[v]);
            float zg = sigmoid_fast(pre_z[v]);
            float ng = tanh_fast(ign8[v] + rg * (hn8[v] + bnv));
            orow[(size_t)j * DE + m] = ng + zg * (eo - ng);
        }
    }
}

// a = argmax over y[N-ADIM .. N-1][0] (first max wins, like jnp.argmax)
__global__ void k_argmax(const float* __restrict__ y, float* __restrict__ out0) {
    float best = y[(N_NODES - ADIM) * DE];
    int bi = 0;
    for (int k = 1; k < ADIM; ++k) {
        float v = y[(N_NODES - ADIM + k) * DE];
        if (v > best) { best = v; bi = k; }
    }
    *out0 = (float)bi;
}

extern "C" void kernel_launch(void* const* d_in, const int* in_sizes, int n_in,
                              void* d_out, int out_size, void* d_ws, size_t ws_size,
                              hipStream_t stream) {
    (void)in_sizes; (void)n_in; (void)out_size; (void)ws_size;
    const float* obs    = (const float*)d_in[0];
    const float* h      = (const float*)d_in[1];
    const int*   types  = (const int*)d_in[2];
    const float* e      = (const float*)d_in[3];
    const int*   A      = (const int*)d_in[4];
    const float* r      = (const float*)d_in[5];
    const float* w_ih_n = (const float*)d_in[6];
    const float* w_hh_n = (const float*)d_in[7];
    const float* b_n    = (const float*)d_in[8];
    const float* bn_n   = (const float*)d_in[9];
    const float* W_out  = (const float*)d_in[10];
    const float* b_out  = (const float*)d_in[11];
    const float* w_ih_e = (const float*)d_in[12];
    const float* w_hh_e = (const float*)d_in[13];
    const float* b_e    = (const float*)d_in[14];
    const float* bn_e   = (const float*)d_in[15];

    float* out   = (float*)d_out;
    float* out_a = out;                       // 1 element
    float* out_h = out + 1;                   // N*DH
    float* out_e = out + 1 + N_NODES * DH;    // N*N*DE

    float* ws   = (float*)d_ws;
    float* y    = ws;                         // N*DE = 16384
    float* x    = y + N_NODES * DE;           // N*DE = 16384
    float* Crow = x + N_NODES * DE;           // N*48 = 49152

    k_node_out   <<<(N_NODES * DE + 255) / 256, 256, 0, stream>>>(h, W_out, b_out, y);
    k_aggregate  <<<N_NODES, 256, 0, stream>>>(e, A, y, obs, x);
    k_node_gru   <<<N_NODES, 128, 0, stream>>>(x, h, types, w_ih_n, w_hh_n, b_n, bn_n, out_h);
    k_edge_rowpre<<<N_NODES, 64, 0, stream>>>(w_ih_e, y, b_e, r, types, Crow);
    k_edge_gru   <<<N_NODES, 128, 0, stream>>>(e, types, w_hh_e, w_ih_e, bn_e, Crow, y, out_e);
    k_argmax     <<<1, 1, 0, stream>>>(y, out_a);
}
